// HSTv4HyperLattice_46291157516393
// MI455X (gfx1250) — matrix operand order repaired and checked by reading, not compiled, not run on hardware
//
#include <hip/hip_runtime.h>
#include <hip/hip_bf16.h>
#include <math.h>

#define B_   4
#define S_   512
#define D_   1024
#define H_   16
#define HD_  64
#define V_   32000
#define LAT_ 48
#define HZ_  8

typedef __attribute__((ext_vector_type(16))) _Float16 v16h;
typedef __attribute__((ext_vector_type(8)))  _Float16 v8h;
typedef __attribute__((ext_vector_type(8)))  float    v8f;

__device__ __forceinline__ float act_fn(float x, int act) {
  if (act == 1) return x > 0.f ? x : 0.f;                                  // relu
  if (act == 2) return 0.5f * x * (1.f + erff(x * 0.7071067811865475f));   // exact gelu
  if (act == 3) return 1.f / (1.f + expf(-x));                             // sigmoid
  return x;
}

// ---------------------------------------------------------------------------
// Weight pre-conversion: f32 -> f16, 8 elements/thread (all weight sizes %8==0)
// ---------------------------------------------------------------------------
__global__ void w_to_f16_kernel(const float* __restrict__ src, _Float16* __restrict__ dst,
                                size_t n8)
{
  size_t i = (size_t)blockIdx.x * blockDim.x + threadIdx.x;
  if (i >= n8) return;
  const float4* s = (const float4*)(src + i * 8);
  float4 x0 = s[0], x1 = s[1];
  v8h h;
  h[0] = (_Float16)x0.x; h[1] = (_Float16)x0.y; h[2] = (_Float16)x0.z; h[3] = (_Float16)x0.w;
  h[4] = (_Float16)x1.x; h[5] = (_Float16)x1.y; h[6] = (_Float16)x1.z; h[7] = (_Float16)x1.w;
  *(v8h*)(dst + i * 8) = h;
}

// ---------------------------------------------------------------------------
// WMMA GEMM: C[m, n] = act( sum_k A[m,k] * Wh[n,k] + bias[n] (+ resid[m,n]) )
// A: [M,K] f32 (activations), Wh: [N,K] f16 (pre-converted weights).
// 128x128 tile, 256 threads = 8 waves; wave owns 32x64 = 8 WMMAs per K-step.
// Interior tiles (uniform branch): B streamed straight into LDS with
// global_load_async_to_lds_b128 (ASYNCcnt), A staged via regs + cvt; one
// barrier per K-step, double-buffered LDS. Edge tiles take a guarded path.
// compute() preloads all 6 fragments so the 8 WMMAs issue back-to-back.
// K % 32 == 0 at every call site.
// ---------------------------------------------------------------------------
__global__ __launch_bounds__(256)
void gemm_wmma_kernel(const float* __restrict__ A, const _Float16* __restrict__ Wh,
                      const float* __restrict__ bias, const float* __restrict__ resid,
                      float* __restrict__ C, int M, int N, int K, int ldc, int act)
{
  constexpr int BM = 128, BN = 128, BK = 32;
  __shared__ __align__(16) _Float16 As[2][BM * BK];
  __shared__ __align__(16) _Float16 Bs[2][BN * BK];

  const int m0   = blockIdx.y * BM;
  const int n0   = blockIdx.x * BN;
  const int tid  = threadIdx.x;
  const int lane = tid & 31;
  const int wave = tid >> 5;
  const int wm   = (wave & 3) * 32;
  const int wn   = (wave >> 2) * 64;

  // A staging: thread covers rows r0, r0+64, 8 contiguous K floats at c0.
  const int r0 = tid >> 2;
  const int c0 = (tid & 3) * 8;
  // B staging: rows br0, br0+64, 8 contiguous K halves at bc0 (16B per chunk).
  const int br0 = tid >> 2;
  const int bc0 = (tid & 3) * 8;

  const bool interior = (m0 + BM <= M) && (n0 + BN <= N);   // uniform per block

  float a_reg[16];

  auto a_load_fast = [&](int k0) {
    const float* p0 = A + (size_t)(m0 + r0) * K + k0 + c0;
    const float* p1 = A + (size_t)(m0 + r0 + 64) * K + k0 + c0;
    *(float4*)(a_reg)      = *(const float4*)p0;
    *(float4*)(a_reg + 4)  = *(const float4*)(p0 + 4);
    *(float4*)(a_reg + 8)  = *(const float4*)p1;
    *(float4*)(a_reg + 12) = *(const float4*)(p1 + 4);
  };
  auto a_load_safe = [&](int k0) {
    if (m0 + r0 < M) {
      const float* p0 = A + (size_t)(m0 + r0) * K + k0 + c0;
      *(float4*)(a_reg)     = *(const float4*)p0;
      *(float4*)(a_reg + 4) = *(const float4*)(p0 + 4);
    } else {
      #pragma unroll
      for (int j = 0; j < 8; ++j) a_reg[j] = 0.f;
    }
    if (m0 + r0 + 64 < M) {
      const float* p1 = A + (size_t)(m0 + r0 + 64) * K + k0 + c0;
      *(float4*)(a_reg + 8)  = *(const float4*)p1;
      *(float4*)(a_reg + 12) = *(const float4*)(p1 + 4);
    } else {
      #pragma unroll
      for (int j = 8; j < 16; ++j) a_reg[j] = 0.f;
    }
  };
  auto cvt8 = [&](const float* s) {
    v8h h;
    #pragma unroll
    for (int j = 0; j < 8; ++j) h[j] = (_Float16)s[j];
    return h;
  };
  auto a_store = [&](int buf) {
    *(v8h*)&As[buf][r0 * BK + c0]        = cvt8(a_reg);
    *(v8h*)&As[buf][(r0 + 64) * BK + c0] = cvt8(a_reg + 8);
  };
  // B tile: async global -> LDS, 16B per lane per op, 2 ops per thread.
  auto b_async = [&](int buf, int k0) {
    {
      const _Float16* gp = Wh + (size_t)(n0 + br0) * K + k0 + bc0;
      unsigned lds = (unsigned)(uintptr_t)&Bs[buf][br0 * BK + bc0];
      asm volatile("global_load_async_to_lds_b128 %0, %1, off"
                   :: "v"(lds), "v"(gp) : "memory");
    }
    {
      const _Float16* gp = Wh + (size_t)(n0 + br0 + 64) * K + k0 + bc0;
      unsigned lds = (unsigned)(uintptr_t)&Bs[buf][(br0 + 64) * BK + bc0];
      asm volatile("global_load_async_to_lds_b128 %0, %1, off"
                   :: "v"(lds), "v"(gp) : "memory");
    }
  };
  auto b_guarded = [&](int buf, int k0) {
    #pragma unroll
    for (int i = 0; i < 2; ++i) {
      int row = br0 + i * 64;
      v8h hv;
      if (n0 + row < N) {
        hv = *(const v8h*)(Wh + (size_t)(n0 + row) * K + k0 + bc0);
      } else {
        #pragma unroll
        for (int j = 0; j < 8; ++j) hv[j] = (_Float16)0.f;
      }
      *(v8h*)&Bs[buf][row * BK + bc0] = hv;
    }
  };
  auto wait_async = [&]() {
    asm volatile("s_wait_asynccnt 0" ::: "memory");
  };

  v8f acc[8];
  #pragma unroll
  for (int i = 0; i < 8; ++i) acc[i] = v8f{};

  // WMMA fragment maps (wave32, 16-bit operands):
  //  A 16x32: lanes 0-15 row=lane, halves 0..7 -> K 0..7, 8..15 -> K 16..23;
  //           lanes16-31 row=lane-16, K offsets +8.
  //  B 32x16: lane c (0-15) col c holds K 0..7,16..23; lane c+16 holds K 8..15,24..31.
  const int ar  = wm + (lane & 15);
  const int akb = (lane >> 4) << 3;
  const int bkb = (lane >> 4) << 3;

  union frag { v16h v; v8h h[2]; };

  auto compute = [&](int buf) {
    const _Float16* as = As[buf];
    const _Float16* bs = Bs[buf];
    frag af[2], bf[4];
    // Preload ALL fragments (12 x ds_load_b128) before the WMMA burst so the
    // eight v_wmma issue back-to-back with a single dscnt wait.
    af[0].h[0] = *(const v8h*)&as[ar * BK + akb];
    af[0].h[1] = *(const v8h*)&as[ar * BK + akb + 16];
    af[1].h[0] = *(const v8h*)&as[(ar + 16) * BK + akb];
    af[1].h[1] = *(const v8h*)&as[(ar + 16) * BK + akb + 16];
    #pragma unroll
    for (int fn = 0; fn < 4; ++fn) {
      const int bc = wn + fn * 16 + (lane & 15);
      bf[fn].h[0] = *(const v8h*)&bs[bc * BK + bkb];
      bf[fn].h[1] = *(const v8h*)&bs[bc * BK + bkb + 16];
    }
    #pragma unroll
    for (int fn = 0; fn < 4; ++fn) {
      acc[fn]     = __builtin_amdgcn_wmma_f32_16x16x32_f16(false, af[0].v, false, bf[fn].v,
                                                           (short)0, acc[fn], false, false);
      acc[4 + fn] = __builtin_amdgcn_wmma_f32_16x16x32_f16(false, af[1].v, false, bf[fn].v,
                                                           (short)0, acc[4 + fn], false, false);
    }
  };

  int buf = 0;
  if (interior) {
    b_async(0, 0);
    a_load_fast(0);
    a_store(0);
    wait_async();
    __syncthreads();
    for (int k0 = BK; k0 < K; k0 += BK) {
      b_async(buf ^ 1, k0);    // async stream next B tile into other buffer
      a_load_fast(k0);         // issue next A tile loads
      compute(buf);            // WMMAs on current buffer hide the latency
      a_store(buf ^ 1);        // cvt + park next A tile
      wait_async();
      __syncthreads();
      buf ^= 1;
    }
    compute(buf);
  } else {
    b_guarded(0, 0);
    a_load_safe(0);
    a_store(0);
    __syncthreads();
    for (int k0 = BK; k0 < K; k0 += BK) {
      b_guarded(buf ^ 1, k0);
      a_load_safe(k0);
      compute(buf);
      a_store(buf ^ 1);
      __syncthreads();
      buf ^= 1;
    }
    compute(buf);
  }

  // Epilogue. C/D frag layout: VGPR v, lanes 0-15 -> M=v, N=lane; lanes16-31 -> M=8+v.
  const int mbase = m0 + wm + ((lane >> 4) << 3);
  const int nbase = n0 + wn + (lane & 15);
  #pragma unroll
  for (int fm = 0; fm < 2; ++fm) {
    #pragma unroll
    for (int fn = 0; fn < 4; ++fn) {
      int n = nbase + fn * 16;
      if (n >= N) continue;
      float bv = bias ? bias[n] : 0.f;
      v8f a = acc[fm * 4 + fn];
      #pragma unroll
      for (int v = 0; v < 8; ++v) {
        int m = mbase + fm * 16 + v;
        if (m >= M) continue;
        float val = a[v] + bv;
        if (resid) val += resid[(size_t)m * N + n];
        C[(size_t)m * ldc + n] = act_fn(val, act);
      }
    }
  }
}

// ---------------------------------------------------------------------------
// Attention: one thread per (b, h, q); online softmax; qkv packed [M, 3*D].
// ---------------------------------------------------------------------------
__global__ void attention_kernel(const float* __restrict__ qkv, float* __restrict__ O,
                                 int causal)
{
  int t = blockIdx.x * blockDim.x + threadIdx.x;
  if (t >= B_ * H_ * S_) return;
  int qi = t % S_;
  int h  = (t / S_) % H_;
  int b  = t / (S_ * H_);
  const size_t D3 = 3 * D_;
  const float4* qp = (const float4*)(qkv + (size_t)(b * S_ + qi) * D3 + h * HD_);
  float4 q[HD_ / 4], acc[HD_ / 4];
  #pragma unroll
  for (int d = 0; d < HD_ / 4; ++d) {
    float4 qv = qp[d];
    q[d] = make_float4(qv.x * 0.125f, qv.y * 0.125f, qv.z * 0.125f, qv.w * 0.125f);
    acc[d] = make_float4(0.f, 0.f, 0.f, 0.f);
  }
  float mx = -3.0e38f, den = 0.f;
  int kend = causal ? (qi + 1) : S_;
  for (int s = 0; s < kend; ++s) {
    const float4* kp = (const float4*)(qkv + (size_t)(b * S_ + s) * D3 + D_ + h * HD_);
    float dot = 0.f;
    #pragma unroll
    for (int d = 0; d < HD_ / 4; ++d) {
      float4 kv = kp[d];
      dot += q[d].x * kv.x + q[d].y * kv.y + q[d].z * kv.z + q[d].w * kv.w;
    }
    float nm   = fmaxf(mx, dot);
    float corr = expf(mx - nm);
    float p    = expf(dot - nm);
    den = den * corr + p;
    const float4* vp = kp + D_ / 4;
    #pragma unroll
    for (int d = 0; d < HD_ / 4; ++d) {
      float4 vv = vp[d];
      acc[d].x = acc[d].x * corr + p * vv.x;
      acc[d].y = acc[d].y * corr + p * vv.y;
      acc[d].z = acc[d].z * corr + p * vv.z;
      acc[d].w = acc[d].w * corr + p * vv.w;
    }
    mx = nm;
  }
  float inv = 1.f / den;
  float4* op = (float4*)(O + (size_t)(b * S_ + qi) * D_ + h * HD_);
  #pragma unroll
  for (int d = 0; d < HD_ / 4; ++d)
    op[d] = make_float4(acc[d].x * inv, acc[d].y * inv, acc[d].z * inv, acc[d].w * inv);
}

// ---------------------------------------------------------------------------
// LayerNorm over D=1024, one block per row, optional fused residual + act.
// ---------------------------------------------------------------------------
__global__ __launch_bounds__(256)
void layernorm_kernel(const float* __restrict__ in, const float* __restrict__ resid,
                      const float* __restrict__ g, const float* __restrict__ b,
                      float* __restrict__ out, int act)
{
  __shared__ float sh[256];
  int row = blockIdx.x;
  int tid = threadIdx.x;
  const float* ip = in + (size_t)row * D_;
  const float* rp = resid ? resid + (size_t)row * D_ : nullptr;
  float v[4];
  float s = 0.f;
  #pragma unroll
  for (int i = 0; i < 4; ++i) {
    int d = tid + i * 256;
    v[i] = ip[d] + (rp ? rp[d] : 0.f);
    s += v[i];
  }
  sh[tid] = s; __syncthreads();
  for (int off = 128; off > 0; off >>= 1) { if (tid < off) sh[tid] += sh[tid + off]; __syncthreads(); }
  float mean = sh[0] * (1.f / D_);
  __syncthreads();
  float s2 = 0.f;
  #pragma unroll
  for (int i = 0; i < 4; ++i) { float dd = v[i] - mean; s2 += dd * dd; }
  sh[tid] = s2; __syncthreads();
  for (int off = 128; off > 0; off >>= 1) { if (tid < off) sh[tid] += sh[tid + off]; __syncthreads(); }
  float rstd = rsqrtf(sh[0] * (1.f / D_) + 1e-5f);
  float* op = out + (size_t)row * D_;
  #pragma unroll
  for (int i = 0; i < 4; ++i) {
    int d = tid + i * 256;
    op[d] = act_fn((v[i] - mean) * rstd * g[d] + b[d], act);
  }
}

// ---------------------------------------------------------------------------
// Elementwise / small kernels
// ---------------------------------------------------------------------------
__global__ void embed_kernel(const int* __restrict__ ids, const float* __restrict__ tok,
                             const float* __restrict__ pos, float* __restrict__ x, int n)
{
  int i = blockIdx.x * blockDim.x + threadIdx.x;
  if (i >= n) return;
  int m = i / D_, d = i % D_;
  int s = m % S_;
  x[i] = tok[(size_t)ids[m] * D_ + d] + pos[(size_t)s * D_ + d];
}

__global__ void lat_adj_kernel(const float* __restrict__ Wlat, float* __restrict__ w,
                               float* __restrict__ has_nb)
{
  int i = threadIdx.x;
  if (i >= LAT_) return;
  const float* rw = Wlat + i * LAT_;
  float mx = -3.0e38f;
  for (int j = 0; j < LAT_; ++j) mx = fmaxf(mx, rw[j]);
  float e[LAT_];
  float s = 0.f;
  for (int j = 0; j < LAT_; ++j) { e[j] = expf(rw[j] - mx); s += e[j]; }
  float inv = 1.f / s;
  float any = 0.f;
  for (int j = 0; j < LAT_; ++j) {
    float p = e[j] * inv;
    bool msk = p > 0.01f;
    w[i * LAT_ + j] = msk ? p : 0.f;
    if (msk) any = 1.f;
  }
  has_nb[i] = any;
}

__global__ void gather_lat_kernel(const float* __restrict__ x, float* __restrict__ xl, int n)
{
  int i = blockIdx.x * blockDim.x + threadIdx.x;
  if (i >= n) return;
  int d = i % D_, r = i / D_;
  int b = r / LAT_, j = r % LAT_;
  xl[i] = x[(size_t)(b * S_ + j) * D_ + d];
}

__global__ void scatter_lat_kernel(const float* __restrict__ xl, float* __restrict__ x, int n)
{
  int i = blockIdx.x * blockDim.x + threadIdx.x;
  if (i >= n) return;
  int d = i % D_, r = i / D_;
  int b = r / LAT_, j = r % LAT_;
  x[(size_t)(b * S_ + j) * D_ + d] = xl[i];
}

__global__ void lat_weighted_kernel(const float* __restrict__ w, const float* __restrict__ xt,
                                    float* __restrict__ out, int n)
{
  int i = blockIdx.x * blockDim.x + threadIdx.x;
  if (i >= n) return;
  int d = i % D_, r = i / D_;
  int b = r / LAT_, ii = r % LAT_;
  float s = 0.f;
  for (int j = 0; j < LAT_; ++j)
    s += w[ii * LAT_ + j] * xt[(size_t)(b * LAT_ + j) * D_ + d];
  out[i] = s;
}

__global__ void concat_kernel(const float* __restrict__ a, const float* __restrict__ b,
                              float* __restrict__ out, int n) // n = rows * 2D
{
  int i = blockIdx.x * blockDim.x + threadIdx.x;
  if (i >= n) return;
  int r = i / (2 * D_), c = i % (2 * D_);
  out[i] = (c < D_) ? a[(size_t)r * D_ + c] : b[(size_t)r * D_ + c - D_];
}

__global__ void lat_select_kernel(const float* __restrict__ msg, const float* __restrict__ xt,
                                  const float* __restrict__ has_nb, float* __restrict__ out, int n)
{
  int i = blockIdx.x * blockDim.x + threadIdx.x;
  if (i >= n) return;
  int ii = (i / D_) % LAT_;
  out[i] = (has_nb[ii] > 0.5f) ? msg[i] : xt[i];
}

__global__ void lat_update_kernel(const float* __restrict__ g, const float* __restrict__ opv,
                                  float* __restrict__ xl, int n)
{
  int i = blockIdx.x * blockDim.x + threadIdx.x;
  if (i >= n) return;
  float gg = g[i];
  xl[i] = gg * opv[i] + (1.f - gg) * xl[i];
}

__global__ void gather_last_kernel(const float* __restrict__ x, float* __restrict__ xl, int n)
{
  int i = blockIdx.x * blockDim.x + threadIdx.x;
  if (i >= n) return;
  int b = i / D_, d = i % D_;
  xl[i] = x[(size_t)(b * S_ + (S_ - 1)) * D_ + d];
}

// ---------------------------------------------------------------------------
// Host orchestration
// ---------------------------------------------------------------------------
extern "C" void kernel_launch(void* const* d_in, const int* in_sizes, int n_in,
                              void* d_out, int out_size, void* d_ws, size_t ws_size,
                              hipStream_t stream)
{
  (void)in_sizes; (void)n_in; (void)out_size; (void)ws_size;
  const int* ids = (const int*)d_in[0];
  auto F = [&](int i) { return (const float*)d_in[i]; };

  float* ws = (float*)d_ws;
  const size_t MD = (size_t)(B_ * S_) * D_;    // 2,097,152
  const size_t LD = (size_t)(B_ * LAT_) * D_;  //   196,608
  float* X    = ws;
  float* X2   = ws + 1 * MD;
  float* NX   = ws + 2 * MD;
  float* BIG  = ws + 3 * MD;   // 4*MD: QKV ([M,3D]) and FFN hidden ([M,4D])
  float* ATT  = ws + 7 * MD;
  float* XLAT = ws + 8 * MD;
  float* XT   = XLAT + LD;
  float* WTD  = XT + LD;
  float* CAT  = WTD + LD;      // 2*LD
  float* E1   = CAT + 2 * LD;
  float* MSG  = E1 + LD;
  float* MSGS = MSG + LD;
  float* GB   = MSGS + LD;
  float* OPV  = GB + LD;
  float* WADJ = OPV + LD;
  float* HNB  = WADJ + LAT_ * LAT_;
  float* XLAST = HNB + 64;
  float* PROJ  = XLAST + B_ * D_;
  float* C1O   = PROJ + B_ * HZ_ * D_;
  // f16 weight buffer (64 MB), 16B aligned, reused per GEMM
  uintptr_t wend = (uintptr_t)(C1O + B_ * (D_ / 4));
  wend = (wend + 15) & ~(uintptr_t)15;
  _Float16* W16 = (_Float16*)wend;

  const int Mtok = B_ * S_;
  auto gemm = [&](const float* A, const float* Wf, const float* bias, const float* resid,
                  float* C, int M, int N, int K, int ldc, int act) {
    size_t n8 = (size_t)N * K / 8;
    w_to_f16_kernel<<<(int)((n8 + 255) / 256), 256, 0, stream>>>(Wf, W16, n8);
    dim3 grid((N + 127) / 128, (M + 127) / 128);
    gemm_wmma_kernel<<<grid, 256, 0, stream>>>(A, W16, bias, resid, C, M, N, K, ldc, act);
  };
  auto ln = [&](const float* in, const float* resid, const float* g, const float* b,
                float* out, int M, int act) {
    layernorm_kernel<<<M, 256, 0, stream>>>(in, resid, g, b, out, act);
  };

  // Embedding
  {
    int n = Mtok * D_;
    embed_kernel<<<(n + 255) / 256, 256, 0, stream>>>(ids, F(1), F(2), X, n);
  }

  // Bottom encoder layers (post-LN, non-causal, ReLU FFN)
  for (int l = 0; l < 2; ++l) {
    int p = 3 + l * 12;  // Wqkv,bqkv,Wo,bo,ln1_g,ln1_b,W1,b1,W2,b2,ln2_g,ln2_b
    gemm(X, F(p + 0), F(p + 1), nullptr, BIG, Mtok, 3 * D_, D_, 3 * D_, 0);  // qkv
    attention_kernel<<<(B_ * H_ * S_ + 255) / 256, 256, 0, stream>>>(BIG, ATT, 0);
    gemm(ATT, F(p + 2), F(p + 3), X, X2, Mtok, D_, D_, D_, 0);               // Wo + resid
    ln(X2, nullptr, F(p + 4), F(p + 5), X, Mtok, 0);                         // ln1
    gemm(X, F(p + 6), F(p + 7), nullptr, BIG, Mtok, 4 * D_, D_, 4 * D_, 1);  // W1 + relu
    gemm(BIG, F(p + 8), F(p + 9), X, X2, Mtok, D_, 4 * D_, D_, 0);           // W2 + resid
    ln(X2, nullptr, F(p + 10), F(p + 11), X, Mtok, 0);                       // ln2
  }

  // Lattice block on x[:, :48]
  {
    const int ML = B_ * LAT_;
    const int nl = ML * D_;
    lat_adj_kernel<<<1, 64, 0, stream>>>(F(27), WADJ, HNB);
    gather_lat_kernel<<<(nl + 255) / 256, 256, 0, stream>>>(X, XLAT, nl);
    gemm(XLAT, F(28), F(29), nullptr, E1, ML, D_, D_, D_, 0);                // nt linear
    ln(E1, nullptr, F(30), F(31), XT, ML, 2);                                // ln + gelu -> xt
    lat_weighted_kernel<<<(nl + 255) / 256, 256, 0, stream>>>(WADJ, XT, WTD, nl);
    concat_kernel<<<(2 * nl + 255) / 256, 256, 0, stream>>>(XT, WTD, CAT, 2 * nl);
    gemm(CAT, F(32), F(33), nullptr, E1, ML, D_, 2 * D_, D_, 2);             // em1 + gelu
    gemm(E1, F(34), F(35), nullptr, MSG, ML, D_, D_, D_, 0);                 // em2
    lat_select_kernel<<<(nl + 255) / 256, 256, 0, stream>>>(MSG, XT, HNB, MSGS, nl);
    concat_kernel<<<(2 * nl + 255) / 256, 256, 0, stream>>>(XLAT, MSGS, CAT, 2 * nl);
    gemm(CAT, F(38), F(39), nullptr, GB, ML, D_, 2 * D_, D_, 3);             // gate + sigmoid
    gemm(MSGS, F(36), F(37), nullptr, OPV, ML, D_, D_, D_, 0);               // op
    lat_update_kernel<<<(nl + 255) / 256, 256, 0, stream>>>(GB, OPV, XLAT, nl);
    scatter_lat_kernel<<<(nl + 255) / 256, 256, 0, stream>>>(XLAT, X, nl);
  }

  // Top decoder layers (pre-LN, causal, GELU FFN)
  for (int l = 0; l < 2; ++l) {
    int p = 40 + l * 16;  // Wq,bq,Wk,bk,Wv,bv,Wo,bo,f1W,f1b,f2W,f2b,n1_g,n1_b,n2_g,n2_b
    ln(X, nullptr, F(p + 12), F(p + 13), NX, Mtok, 0);                       // n1
    gemm(NX, F(p + 0), F(p + 1), nullptr, BIG + 0,      Mtok, D_, D_, 3 * D_, 0);
    gemm(NX, F(p + 2), F(p + 3), nullptr, BIG + D_,     Mtok, D_, D_, 3 * D_, 0);
    gemm(NX, F(p + 4), F(p + 5), nullptr, BIG + 2 * D_, Mtok, D_, D_, 3 * D_, 0);
    attention_kernel<<<(B_ * H_ * S_ + 255) / 256, 256, 0, stream>>>(BIG, ATT, 1);
    gemm(ATT, F(p + 6), F(p + 7), X, X2, Mtok, D_, D_, D_, 0);               // Wo + resid
    ln(X2, nullptr, F(p + 14), F(p + 15), NX, Mtok, 0);                      // n2
    gemm(NX, F(p + 8), F(p + 9), nullptr, BIG, Mtok, 4 * D_, D_, 4 * D_, 2); // f1 + gelu
    gemm(BIG, F(p + 10), F(p + 11), X2, X, Mtok, D_, 4 * D_, D_, 0);         // f2 + resid
  }

  // Final LN + heads
  float* logits  = (float*)d_out;                       // [B,S,V]
  float* horizon = logits + (size_t)Mtok * V_;          // [B,HZ,V]
  float* conf    = horizon + (size_t)B_ * HZ_ * V_;     // [B,HZ]
  ln(X, nullptr, F(72), F(73), NX, Mtok, 0);
  gemm(NX, F(74), nullptr, nullptr, logits, Mtok, V_, D_, V_, 0);            // lm head

  gather_last_kernel<<<(B_ * D_ + 255) / 256, 256, 0, stream>>>(X, XLAST, B_ * D_);
  gemm(XLAST, F(75), F(76), nullptr, PROJ, B_, HZ_ * D_, D_, HZ_ * D_, 0);   // proj
  gemm(PROJ, F(77), nullptr, nullptr, horizon, B_ * HZ_, V_, D_, V_, 0);     // pred_W
  gemm(XLAST, F(78), F(79), nullptr, C1O, B_, D_ / 4, D_, D_ / 4, 1);        // c1 + relu
  gemm(C1O, F(80), F(81), nullptr, conf, B_, HZ_, D_ / 4, HZ_, 3);           // c2 + sigmoid
}
